// IntraModule3_80169859547840
// MI455X (gfx1250) — matrix operand order repaired and checked by reading, not compiled, not run on hardware
//
#include <hip/hip_runtime.h>

#define B_ 2
#define NFRAMES_ 3
#define C_ 64
#define GC_ 192
#define RC_ 12
#define H_ 96
#define W_ 96
#define HW_ 9216
#define TQ 16
#define TK 32

typedef __attribute__((ext_vector_type(16))) __bf16 v16bf;
typedef __attribute__((ext_vector_type(8)))  float  v8f;

// ---------------------------------------------------------------------------
// conv1x1 stage 1: x[b,o,hw] = b_cg1[o] + sum_ic w_cg1[o,ic] * group[b,ic,hw]
__global__ __launch_bounds__(256) void k_cg1(const float* __restrict__ grp,
    const float* __restrict__ w, const float* __restrict__ bias, float* __restrict__ x) {
  int hw = blockIdx.x * 256 + threadIdx.x;
  int o = blockIdx.y, b = blockIdx.z;
  const float* in = grp + (size_t)b * GC_ * HW_ + hw;
  const float* wr = w + o * GC_;
  float acc = bias[o];
  for (int ic = 0; ic < GC_; ++ic) acc += wr[ic] * in[(size_t)ic * HW_];
  x[((size_t)b * GC_ + o) * HW_ + hw] = acc;
}

// y[b,o] = mean over HW of x
__global__ __launch_bounds__(256) void k_rowmean(const float* __restrict__ x, float* __restrict__ y) {
  __shared__ float red[256];
  int o = blockIdx.x, b = blockIdx.y, tid = threadIdx.x;
  const float* p = x + ((size_t)b * GC_ + o) * HW_;
  float acc = 0.f;
  for (int i = tid; i < HW_; i += 256) acc += p[i];
  red[tid] = acc; __syncthreads();
  for (int s = 128; s; s >>= 1) { if (tid < s) red[tid] += red[tid + s]; __syncthreads(); }
  if (tid == 0) y[b * GC_ + o] = red[0] * (1.0f / HW_);
}

// squeeze-excite: s = sigmoid(relu(y@w1^T+b1)@w2^T+b2)
__global__ __launch_bounds__(256) void k_se(const float* __restrict__ y,
    const float* __restrict__ w1, const float* __restrict__ b1,
    const float* __restrict__ w2, const float* __restrict__ b2, float* __restrict__ s) {
  __shared__ float yv[GC_];
  __shared__ float hid[RC_];
  int b = blockIdx.x, tid = threadIdx.x;
  if (tid < GC_) yv[tid] = y[b * GC_ + tid];
  __syncthreads();
  if (tid < RC_) {
    float a = b1[tid];
    for (int i = 0; i < GC_; ++i) a += w1[tid * GC_ + i] * yv[i];
    hid[tid] = fmaxf(a, 0.f);
  }
  __syncthreads();
  if (tid < GC_) {
    float a = b2[tid];
    for (int r = 0; r < RC_; ++r) a += w2[tid * RC_ + r] * hid[r];
    s[b * GC_ + tid] = 1.f / (1.f + __expf(-a));
  }
}

// conv1x1 with SE scale: x2[b,c,hw] = b_cg2[c] + sum_o w_cg2[c,o]*s[b,o]*x[b,o,hw]
__global__ __launch_bounds__(256) void k_cg2(const float* __restrict__ x,
    const float* __restrict__ s, const float* __restrict__ w,
    const float* __restrict__ bias, float* __restrict__ x2) {
  int hw = blockIdx.x * 256 + threadIdx.x;
  int c = blockIdx.y, b = blockIdx.z;
  const float* in = x + (size_t)b * GC_ * HW_ + hw;
  const float* wr = w + c * GC_;
  const float* sr = s + b * GC_;
  float acc = bias[c];
  for (int o = 0; o < GC_; ++o) acc += wr[o] * sr[o] * in[(size_t)o * HW_];
  x2[((size_t)b * C_ + c) * HW_ + hw] = acc;
}

// conv1x1 64x64: fea = w_cg3 @ x2 + b_cg3
__global__ __launch_bounds__(256) void k_cg3(const float* __restrict__ x2,
    const float* __restrict__ w, const float* __restrict__ bias, float* __restrict__ fea) {
  int hw = blockIdx.x * 256 + threadIdx.x;
  int c = blockIdx.y, b = blockIdx.z;
  const float* in = x2 + (size_t)b * C_ * HW_ + hw;
  const float* wr = w + c * C_;
  float acc = bias[c];
  for (int ci = 0; ci < C_; ++ci) acc += wr[ci] * in[(size_t)ci * HW_];
  fea[((size_t)b * C_ + c) * HW_ + hw] = acc;
}

// projections Ff,Gf (from fea) and Hf, Hf^2 (from BG center frame), packed bf16
__global__ __launch_bounds__(256) void k_proj(const float* __restrict__ fea,
    const float* __restrict__ BG,
    const float* __restrict__ wf, const float* __restrict__ bf,
    const float* __restrict__ wg, const float* __restrict__ bg,
    const float* __restrict__ wh, const float* __restrict__ bh,
    __bf16* __restrict__ Ff, __bf16* __restrict__ Gf,
    __bf16* __restrict__ Hf, __bf16* __restrict__ Hf2) {
  int hw = blockIdx.x * 256 + threadIdx.x;
  int co = blockIdx.y, b = blockIdx.z;
  const float* fb = fea + (size_t)b * C_ * HW_ + hw;
  const float* gb = BG + ((size_t)(b * NFRAMES_ + 1) * C_) * HW_ + hw;
  float aF = bf[co], aG = bg[co], aH = bh[co];
  for (int ci = 0; ci < C_; ++ci) {
    float f = fb[(size_t)ci * HW_];
    aF += wf[co * C_ + ci] * f;
    aG += wg[co * C_ + ci] * f;
    float h = gb[(size_t)ci * HW_];
    aH += wh[co * C_ + ci] * h;
  }
  size_t idx = ((size_t)b * C_ + co) * HW_ + hw;
  Ff[idx] = (__bf16)aF;
  Gf[idx] = (__bf16)aG;
  Hf[idx] = (__bf16)aH;
  Hf2[idx] = (__bf16)(aH * aH);
}

// ---------------------------------------------------------------------------
// Flash-attention with dual value streams (Hf and Hf^2), bf16 WMMA, f32 accum.
// One wave owns 16 queries; 4 waves/block share K/V LDS tiles of 32 keys.
__global__ __launch_bounds__(128) void k_attn(
    const __bf16* __restrict__ Qg, const __bf16* __restrict__ Kg,
    const __bf16* __restrict__ Vg, const __bf16* __restrict__ V2g,
    float* __restrict__ meanA, float* __restrict__ stdA) {
  __shared__ __align__(16) __bf16 Ks[TK * C_];       // [j][c] (transposed)
  __shared__ __align__(16) __bf16 Vs[C_ * TK];       // [c][j]
  __shared__ __align__(16) __bf16 V2s[C_ * TK];      // [c][j]
  __shared__ __align__(16) __bf16 Plds[4][TQ * TK];  // per-wave P scratch [m][j]

  const int tid  = threadIdx.x;
  const int wid  = tid >> 5;
  const int lane = tid & 31;
  const int g    = lane >> 4;   // lane group (ISA 7.12.2)
  const int ln   = lane & 15;
  const int b    = blockIdx.y;
  const int i0   = (blockIdx.x * 4 + wid) * TQ;

  const __bf16* Qb  = Qg  + (size_t)b * C_ * HW_;
  const __bf16* Kb  = Kg  + (size_t)b * C_ * HW_;
  const __bf16* Vb  = Vg  + (size_t)b * C_ * HW_;
  const __bf16* V2b = V2g + (size_t)b * C_ * HW_;

  // Q A-fragments (16-bit A 16x32 layout), c-chunks 0 and 32
  v16bf qa[2];
#pragma unroll
  for (int cc = 0; cc < 2; ++cc)
#pragma unroll
    for (int e = 0; e < 16; ++e) {
      int v = e >> 1, h = e & 1;
      int k = (g << 4) + e;
      qa[cc][e] = Qb[(size_t)(cc * 32 + k) * HW_ + i0 + ln];
    }

  v8f om[4] = {};  // mean accumulators, N chunks 0/16/32/48
  v8f o2[4] = {};  // m2 accumulators
  float mrow[8], lrow[8];
#pragma unroll
  for (int r = 0; r < 8; ++r) { mrow[r] = -1e30f; lrow[r] = 0.f; }

  const int crow  = tid >> 1;  // 0..63 channel row for cooperative loads
  const int chalf = tid & 1;   // half of the 32-key tile

  for (int js = 0; js < HW_ / TK; ++js) {
    const int j0 = js * TK;
    __syncthreads();
    {
      // K tile: transpose-on-store -> Ks[j][c] (B-frag gathers become b128)
      const __bf16* sk = Kb + (size_t)crow * HW_ + j0 + chalf * 16;
#pragma unroll
      for (int jj = 0; jj < 16; ++jj)
        Ks[(chalf * 16 + jj) * C_ + crow] = sk[jj];
      // V / V^2 tiles stay [c][j]: contiguous 128-bit copies
      const uint4* s2 = (const uint4*)(Vb  + (size_t)crow * HW_ + j0 + chalf * 16);
      const uint4* s3 = (const uint4*)(V2b + (size_t)crow * HW_ + j0 + chalf * 16);
      uint4* d2 = (uint4*)(Vs  + crow * TK + chalf * 16);
      uint4* d3 = (uint4*)(V2s + crow * TK + chalf * 16);
      d2[0] = s2[0]; d2[1] = s2[1];
      d3[0] = s3[0]; d3[1] = s3[1];
    }
    __syncthreads();

    // scores: 16 queries x 32 keys, K-dim = 64 channels (2 WMMAs per 16-key tile)
    v8f sc[2] = {};
#pragma unroll
    for (int jt = 0; jt < 2; ++jt)
#pragma unroll
      for (int cc = 0; cc < 2; ++cc) {
        v16bf bk;
#pragma unroll
        for (int e = 0; e < 16; ++e)
          bk[e] = Ks[(jt * 16 + ln) * C_ + cc * 32 + g * 16 + e];
        sc[jt] = __builtin_amdgcn_wmma_f32_16x16x32_bf16(
            false, qa[cc], false, bk, (short)0, sc[jt], false, false);
      }

    // online softmax (C/D layout: row = r + 8*g striped over 16 lanes of group g)
#pragma unroll
    for (int r = 0; r < 8; ++r) {
      float t = fmaxf(sc[0][r], sc[1][r]);
      t = fmaxf(t, __shfl_xor(t, 1, 16));
      t = fmaxf(t, __shfl_xor(t, 2, 16));
      t = fmaxf(t, __shfl_xor(t, 4, 16));
      t = fmaxf(t, __shfl_xor(t, 8, 16));
      float mnew  = fmaxf(mrow[r], t);
      float alpha = __expf(mrow[r] - mnew);
      float p0 = __expf(sc[0][r] - mnew);
      float p1 = __expf(sc[1][r] - mnew);
      sc[0][r] = p0; sc[1][r] = p1;
      float rs = p0 + p1;
      rs += __shfl_xor(rs, 1, 16);
      rs += __shfl_xor(rs, 2, 16);
      rs += __shfl_xor(rs, 4, 16);
      rs += __shfl_xor(rs, 8, 16);
      lrow[r] = lrow[r] * alpha + rs;
      mrow[r] = mnew;
#pragma unroll
      for (int f = 0; f < 4; ++f) { om[f][r] *= alpha; o2[f][r] *= alpha; }
    }

    // P (C/D layout) -> per-wave LDS, then regather as A-fragment
    __bf16* P = Plds[wid];
#pragma unroll
    for (int r = 0; r < 8; ++r) {
      P[(r + 8 * g) * TK + ln]      = (__bf16)sc[0][r];
      P[(r + 8 * g) * TK + 16 + ln] = (__bf16)sc[1][r];
    }
    asm volatile("s_wait_dscnt 0" ::: "memory");

    v16bf pa;
#pragma unroll
    for (int e = 0; e < 16; ++e) {
      int v = e >> 1, h = e & 1;
      int k = (g << 4) + e;
      pa[e] = P[ln * TK + k];
    }

    // O += P@V ; O2 += P@V^2 (N = 64 in 4 chunks of 16)
#pragma unroll
    for (int f = 0; f < 4; ++f) {
      v16bf bv, bw;
#pragma unroll
      for (int e = 0; e < 16; ++e) {
        bv[e] = Vs [(f * 16 + ln) * TK + g * 16 + e];
        bw[e] = V2s[(f * 16 + ln) * TK + g * 16 + e];
      }
      om[f] = __builtin_amdgcn_wmma_f32_16x16x32_bf16(false, pa, false, bv, (short)0, om[f], false, false);
      o2[f] = __builtin_amdgcn_wmma_f32_16x16x32_bf16(false, pa, false, bw, (short)0, o2[f], false, false);
    }
  }

  // finalize: mean and std per (c, i)
#pragma unroll
  for (int f = 0; f < 4; ++f)
#pragma unroll
    for (int r = 0; r < 8; ++r) {
      float inv = 1.0f / lrow[r];
      int i = i0 + r + 8 * g;
      int c = f * 16 + ln;
      float mn = om[f][r] * inv;
      float m2 = o2[f][r] * inv;
      size_t oi = ((size_t)b * C_ + c) * HW_ + i;
      meanA[oi] = mn;
      stdA[oi]  = sqrtf(fmaxf(m2 - mn * mn, 0.f));
    }
}

// ---------------------------------------------------------------------------
// per-(b,c) mean/var of FG center frame
__global__ __launch_bounds__(256) void k_fgstats(const float* __restrict__ FG,
    float* __restrict__ fgm, float* __restrict__ fgv) {
  __shared__ float rs[256], rq[256];
  int c = blockIdx.x, b = blockIdx.y, tid = threadIdx.x;
  const float* p = FG + ((size_t)(b * NFRAMES_ + 1) * C_ + c) * HW_;
  float s = 0.f, q = 0.f;
  for (int i = tid; i < HW_; i += 256) { float v = p[i]; s += v; q += v * v; }
  rs[tid] = s; rq[tid] = q; __syncthreads();
  for (int t = 128; t; t >>= 1) {
    if (tid < t) { rs[tid] += rs[tid + t]; rq[tid] += rq[tid + t]; }
    __syncthreads();
  }
  if (tid == 0) {
    float m = rs[0] * (1.0f / HW_);
    fgm[b * C_ + c] = m;
    fgv[b * C_ + c] = rq[0] * (1.0f / HW_) - m * m;
  }
}

// xsum = std * mvn(FGc) + mean + BGc
__global__ __launch_bounds__(256) void k_xsum(const float* __restrict__ FG,
    const float* __restrict__ BG, const float* __restrict__ fgm,
    const float* __restrict__ fgv, const float* __restrict__ meanA,
    const float* __restrict__ stdA, float* __restrict__ xsum) {
  int idx = blockIdx.x * 256 + threadIdx.x;
  int b = idx / (C_ * HW_);
  int rem = idx - b * C_ * HW_;
  int c = rem / HW_;
  int hw = rem - c * HW_;
  size_t fgi = ((size_t)(b * NFRAMES_ + 1) * C_ + c) * HW_ + hw;
  float fg = FG[fgi], bg = BG[fgi];
  float m = fgm[b * C_ + c], v = fgv[b * C_ + c];
  float mvn = (fg - m) * rsqrtf(v + 1e-5f);
  xsum[idx] = stdA[idx] * mvn + meanA[idx] + bg;
}

// direct 3x3 SAME conv; mode 0: relu, mode 1: out = mulsrc * sigmoid(conv)
__global__ __launch_bounds__(256) void k_conv3(const float* __restrict__ in,
    const float* __restrict__ w, const float* __restrict__ bias,
    const float* __restrict__ mulsrc, float* __restrict__ out, int mode) {
  int hw = blockIdx.x * 256 + threadIdx.x;
  int co = blockIdx.y, b = blockIdx.z;
  int h = hw / W_, wc = hw - h * W_;
  float acc = bias[co];
  const float* ib = in + (size_t)b * C_ * HW_;
  const float* wr = w + co * C_ * 9;
  for (int ci = 0; ci < C_; ++ci) {
    const float* pc = ib + (size_t)ci * HW_;
    const float* wk = wr + ci * 9;
#pragma unroll
    for (int dy = -1; dy <= 1; ++dy) {
      int hh = h + dy;
      if ((unsigned)hh >= (unsigned)H_) continue;
#pragma unroll
      for (int dx = -1; dx <= 1; ++dx) {
        int ww = wc + dx;
        if ((unsigned)ww >= (unsigned)W_) continue;
        acc += wk[(dy + 1) * 3 + (dx + 1)] * pc[hh * W_ + ww];
      }
    }
  }
  size_t oi = ((size_t)b * C_ + co) * HW_ + hw;
  out[oi] = (mode == 0) ? fmaxf(acc, 0.f)
                        : mulsrc[oi] * (1.0f / (1.0f + __expf(-acc)));
}

// ---------------------------------------------------------------------------
extern "C" void kernel_launch(void* const* d_in, const int* in_sizes, int n_in,
                              void* d_out, int out_size, void* d_ws, size_t ws_size,
                              hipStream_t stream) {
  (void)in_sizes; (void)n_in; (void)out_size; (void)ws_size;
  const float* group = (const float*)d_in[0];
  const float* FG    = (const float*)d_in[1];
  const float* BG    = (const float*)d_in[2];
  const float* w_cg1 = (const float*)d_in[3];
  const float* b_cg1 = (const float*)d_in[4];
  const float* w_ca1 = (const float*)d_in[5];
  const float* b_ca1 = (const float*)d_in[6];
  const float* w_ca2 = (const float*)d_in[7];
  const float* b_ca2 = (const float*)d_in[8];
  const float* w_cg2 = (const float*)d_in[9];
  const float* b_cg2 = (const float*)d_in[10];
  const float* w_cg3 = (const float*)d_in[11];
  const float* b_cg3 = (const float*)d_in[12];
  const float* w_f   = (const float*)d_in[13];
  const float* b_f   = (const float*)d_in[14];
  const float* w_g   = (const float*)d_in[15];
  const float* b_g   = (const float*)d_in[16];
  const float* w_h   = (const float*)d_in[17];
  const float* b_h   = (const float*)d_in[18];
  const float* w_a1  = (const float*)d_in[19];
  const float* b_a1  = (const float*)d_in[20];
  const float* w_a2  = (const float*)d_in[21];
  const float* b_a2  = (const float*)d_in[22];
  float* out = (float*)d_out;

  char* ws = (char*)d_ws;
  size_t off = 0;
  auto alloc = [&](size_t bytes) -> void* {
    off = (off + 255) & ~(size_t)255;
    void* p = ws + off;
    off += bytes;
    return p;
  };

  float*  x     = (float*)alloc((size_t)B_ * GC_ * HW_ * 4);  // reused later as a1
  float*  y     = (float*)alloc((size_t)B_ * GC_ * 4);
  float*  sgl   = (float*)alloc((size_t)B_ * GC_ * 4);
  float*  x2    = (float*)alloc((size_t)B_ * C_ * HW_ * 4);
  float*  fea   = (float*)alloc((size_t)B_ * C_ * HW_ * 4);   // reused later as xsum
  __bf16* Ff    = (__bf16*)alloc((size_t)B_ * C_ * HW_ * 2);
  __bf16* Gf    = (__bf16*)alloc((size_t)B_ * C_ * HW_ * 2);
  __bf16* Hf    = (__bf16*)alloc((size_t)B_ * C_ * HW_ * 2);
  __bf16* Hf2   = (__bf16*)alloc((size_t)B_ * C_ * HW_ * 2);
  float*  meanA = (float*)alloc((size_t)B_ * C_ * HW_ * 4);
  float*  stdA  = (float*)alloc((size_t)B_ * C_ * HW_ * 4);
  float*  fgm   = (float*)alloc((size_t)B_ * C_ * 4);
  float*  fgv   = (float*)alloc((size_t)B_ * C_ * 4);
  float*  xsum  = fea;  // fea is dead after k_proj
  float*  a1    = x;    // x is dead after k_cg2

  dim3 blk(256);
  k_cg1    <<<dim3(HW_ / 256, GC_, B_), blk, 0, stream>>>(group, w_cg1, b_cg1, x);
  k_rowmean<<<dim3(GC_, B_),            blk, 0, stream>>>(x, y);
  k_se     <<<dim3(B_),                 blk, 0, stream>>>(y, w_ca1, b_ca1, w_ca2, b_ca2, sgl);
  k_cg2    <<<dim3(HW_ / 256, C_, B_),  blk, 0, stream>>>(x, sgl, w_cg2, b_cg2, x2);
  k_cg3    <<<dim3(HW_ / 256, C_, B_),  blk, 0, stream>>>(x2, w_cg3, b_cg3, fea);
  k_proj   <<<dim3(HW_ / 256, C_, B_),  blk, 0, stream>>>(fea, BG, w_f, b_f, w_g, b_g, w_h, b_h,
                                                          Ff, Gf, Hf, Hf2);
  k_attn   <<<dim3(HW_ / (4 * TQ), B_), dim3(128), 0, stream>>>(Ff, Gf, Hf, Hf2, meanA, stdA);
  k_fgstats<<<dim3(C_, B_),             blk, 0, stream>>>(FG, fgm, fgv);
  k_xsum   <<<dim3((B_ * C_ * HW_) / 256), blk, 0, stream>>>(FG, BG, fgm, fgv, meanA, stdA, xsum);
  k_conv3  <<<dim3(HW_ / 256, C_, B_),  blk, 0, stream>>>(xsum, w_a1, b_a1, nullptr, a1, 0);
  k_conv3  <<<dim3(HW_ / 256, C_, B_),  blk, 0, stream>>>(a1, w_a2, b_a2, xsum, out, 1);
}
